// EdgeEncoder_49220325212323
// MI455X (gfx1250) — matrix operand order repaired and checked by reading, not compiled, not run on hardware
//
#include <hip/hip_runtime.h>
#include <hip/hip_bf16.h>

#define ZB 4
#define NN 2048
#define KK 16
#define DM 256
#define EE (ZB * NN * KK)   // 131072 edges

typedef unsigned short u16x8 __attribute__((ext_vector_type(8)));
typedef __bf16 v16bf __attribute__((ext_vector_type(16)));
typedef float v8f __attribute__((ext_vector_type(8)));

union V16 {
    struct { u16x8 lo; u16x8 hi; } s;
    v16bf v;
};

__device__ __forceinline__ unsigned short f32_to_bf16(float f) {
    union { float f; unsigned u; } x; x.f = f;
    unsigned r = x.u + 0x7FFFu + ((x.u >> 16) & 1u);   // round-to-nearest-even
    return (unsigned short)(r >> 16);
}
__device__ __forceinline__ float bf16_to_f32(unsigned short h) {
    union { unsigned u; float f; } x; x.u = ((unsigned)h) << 16;
    return x.f;
}
__device__ __forceinline__ float wred(float v) {
#pragma unroll
    for (int off = 16; off > 0; off >>= 1) v += __shfl_xor(v, off, 32);
    return v;
}

// ---------------------------------------------------------------------------
// Weight pack: f32 [Kdim,256] row-major -> bf16 WMMA B-tile layout.
// Layout: entry t = ((kt*16 + nT)*32 + lane), 16 bf16 per entry.
// lane: N = nT*16 + (lane&15); K = kt*32 + (lane>>4)*16 + j  (j=0..15)
// ---------------------------------------------------------------------------
__global__ void pack_w_kernel(const float* __restrict__ W,
                              unsigned short* __restrict__ out, int Kdim) {
    int t = blockIdx.x * blockDim.x + threadIdx.x;
    int total = (Kdim >> 5) * 16 * 32;
    if (t >= total) return;
    int lane = t & 31;
    int rest = t >> 5;
    int nT = rest & 15;
    int kt = rest >> 4;
    int n = nT * 16 + (lane & 15);
    int kBase = kt * 32 + (lane >> 4) * 16;
    unsigned short* dst = out + (size_t)t * 16;
#pragma unroll
    for (int j = 0; j < 16; ++j)
        dst[j] = f32_to_bf16(W[(size_t)(kBase + j) * DM + n]);
}

// ---------------------------------------------------------------------------
// kNN: one thread per (z,n); insertion top-16 on squared CA distance.
// ---------------------------------------------------------------------------
__global__ void knn_kernel(const float* __restrict__ coords,
                           const unsigned char* __restrict__ valid,
                           int* __restrict__ nbrsOut, int* __restrict__ maskOut) {
    int t = blockIdx.x * blockDim.x + threadIdx.x;
    if (t >= ZB * NN) return;
    int z = t / NN, n = t % NN;
    const float* base = coords + (size_t)z * NN * 12;  // [N,4,3]
    float qx = base[n * 12 + 3], qy = base[n * 12 + 4], qz = base[n * 12 + 5]; // CA
    int vq = valid[z * NN + n];
    float bd[16]; int bi[16];
#pragma unroll
    for (int i = 0; i < 16; ++i) { bd[i] = 3.0e38f; bi[i] = n; }
    for (int j = 0; j < NN; ++j) {
        float dx = qx - base[j * 12 + 3];
        float dy = qy - base[j * 12 + 4];
        float dz = qz - base[j * 12 + 5];
        float d2 = dx * dx + dy * dy + dz * dz;
        int vj = valid[z * NN + j];
        if (!(vq | vj)) d2 = 3.0e38f;
        if (d2 < bd[15]) {
            bd[15] = d2; bi[15] = j;
#pragma unroll
            for (int s = 15; s > 0; --s) {
                if (bd[s] < bd[s - 1]) {
                    float td = bd[s]; bd[s] = bd[s - 1]; bd[s - 1] = td;
                    int ti = bi[s]; bi[s] = bi[s - 1]; bi[s - 1] = ti;
                }
            }
        }
    }
#pragma unroll
    for (int k = 0; k < 16; ++k) {
        int m = vq && valid[z * NN + bi[k]];
        nbrsOut[t * 16 + k] = m ? bi[k] : n;
        maskOut[t * 16 + k] = m;
    }
}

// ---------------------------------------------------------------------------
// Per-edge features: one wave32 per edge.
//   - 16 atom-pair dists -> 256 RBF -> LN -> bf16 A_rbf [E,256]
//   - rel frames (F^T Fnbr, 9) @ frame_w + frame_b -> bf16 Hcat[:,256:512]
//   - rel seq embedding gather -> bf16 Hcat[:,512:768]
// ---------------------------------------------------------------------------
__global__ void feat_kernel(const float* __restrict__ coords,
                            const float* __restrict__ frames,
                            const int* __restrict__ seq_pos,
                            const int* __restrict__ chain_pos,
                            const int* __restrict__ nbrs,
                            const float* __restrict__ g_rbf,
                            const float* __restrict__ b_rbf,
                            const float* __restrict__ frame_w,
                            const float* __restrict__ frame_b,
                            const float* __restrict__ seq_emb,
                            unsigned short* __restrict__ A_rbf,
                            unsigned short* __restrict__ Hcat) {
    int gid = blockIdx.x * blockDim.x + threadIdx.x;
    int e = gid >> 5;
    int lane = gid & 31;
    if (e >= EE) return;
    int z = e / (NN * KK);
    int rem = e % (NN * KK);
    int n = rem / KK;
    int nb = nbrs[e];

    // --- RBF: lane covers atom-pair p=lane>>1, 8 rbf centers r0=(lane&1)*8 ---
    int p = lane >> 1, ai = p >> 2, aj = p & 3;
    const float* cs = coords + ((size_t)(z * NN + n) * 4 + ai) * 3;
    const float* cn = coords + ((size_t)(z * NN + nb) * 4 + aj) * 3;
    float dx = cs[0] - cn[0], dy = cs[1] - cn[1], dz = cs[2] - cn[2];
    float d = sqrtf(dx * dx + dy * dy + dz * dz);
    float x[8];
    float s = 0.f, sq = 0.f;
    int r0 = (lane & 1) * 8;
#pragma unroll
    for (int j = 0; j < 8; ++j) {
        float c = 2.f + (float)(r0 + j) * (20.f / 15.f);   // linspace(2,22,16)
        float t = d - c;
        float v = __expf(-(t * t) * (1.f / 1.5625f));       // SPREAD^2 = 1.25^2
        x[j] = v; s += v; sq += v * v;
    }
    s = wred(s); sq = wred(sq);
    float mean = s * (1.f / 256.f);
    float var = sq * (1.f / 256.f) - mean * mean;
    float inv = rsqrtf(var + 1e-5f);
    unsigned short* ar = A_rbf + (size_t)e * 256 + lane * 8;
#pragma unroll
    for (int j = 0; j < 8; ++j) {
        int f = lane * 8 + j;
        ar[j] = f32_to_bf16((x[j] - mean) * inv * g_rbf[f] + b_rbf[f]);
    }

    // --- relative frames: rel[jj*3+ll] = sum_i F[i][jj]*Fn[i][ll] ---
    const float* F = frames + (size_t)(z * NN + n) * 9;
    const float* Fn = frames + (size_t)(z * NN + nb) * 9;
    float r9[9];
#pragma unroll
    for (int jj = 0; jj < 3; ++jj)
#pragma unroll
        for (int ll = 0; ll < 3; ++ll) {
            float a = 0.f;
#pragma unroll
            for (int i = 0; i < 3; ++i) a += F[i * 3 + jj] * Fn[i * 3 + ll];
            r9[jj * 3 + ll] = a;
        }
    unsigned short* hf = Hcat + (size_t)e * 768 + 256 + lane * 8;
#pragma unroll
    for (int j = 0; j < 8; ++j) {
        int c = lane * 8 + j;
        float a = frame_b[c];
#pragma unroll
        for (int q = 0; q < 9; ++q) a += r9[q] * frame_w[q * DM + c];
        hf[j] = f32_to_bf16(a);
    }

    // --- relative seq embedding ---
    int sp = seq_pos[z * NN + n], spn = seq_pos[z * NN + nb];
    int cp = chain_pos[z * NN + n], cpn = chain_pos[z * NN + nb];
    int rel = spn - sp;
    rel = rel < -32 ? -32 : (rel > 32 ? 32 : rel);
    if (cp != cpn) rel = 33;
    int idx = rel + 32;
    unsigned short* hs = Hcat + (size_t)e * 768 + 512 + lane * 8;
#pragma unroll
    for (int j = 0; j < 8; ++j)
        hs[j] = f32_to_bf16(seq_emb[idx * DM + lane * 8 + j]);
}

// ---------------------------------------------------------------------------
// Edge LayerNorm over 768 (bf16 in/out): one wave per edge, 24 feats/lane.
// ---------------------------------------------------------------------------
__global__ void edgeln_kernel(const unsigned short* __restrict__ Hcat,
                              const float* __restrict__ g,
                              const float* __restrict__ b,
                              unsigned short* __restrict__ Xln) {
    int gid = blockIdx.x * blockDim.x + threadIdx.x;
    int e = gid >> 5;
    int lane = gid & 31;
    if (e >= EE) return;
    const unsigned short* h = Hcat + (size_t)e * 768 + lane * 24;
    float x[24];
    float s = 0.f, sq = 0.f;
#pragma unroll
    for (int j = 0; j < 24; ++j) {
        float v = bf16_to_f32(h[j]);
        x[j] = v; s += v; sq += v * v;
    }
    s = wred(s); sq = wred(sq);
    float mean = s * (1.f / 768.f);
    float var = sq * (1.f / 768.f) - mean * mean;
    float inv = rsqrtf(var + 1e-5f);
    unsigned short* o = Xln + (size_t)e * 768 + lane * 24;
#pragma unroll
    for (int j = 0; j < 24; ++j) {
        int f = lane * 24 + j;
        o[j] = f32_to_bf16((x[j] - mean) * inv * g[f] + b[f]);
    }
}

// ---------------------------------------------------------------------------
// WMMA GEMM: [M,Kdim]bf16 (row-major) @ packed-W[Kdim,256]bf16 + bias.
// One wave per 16x16 output tile; K swept in steps of 32 via
// v_wmma_f32_16x16x32_bf16. ACT=1 -> SiLU. OUTF32=1 -> f32 out else bf16.
// A lane layout contiguous: lane<16 K{0..15},
// lane>=16 K{16..31}. B packed with half-wave K split (see pack_w).
// ---------------------------------------------------------------------------
template <int ACT, int OUTF32>
__global__ void gemm_bf16_kernel(const unsigned short* __restrict__ A,
                                 const unsigned short* __restrict__ Bp,
                                 const float* __restrict__ bias,
                                 unsigned short* __restrict__ outB,
                                 float* __restrict__ outF,
                                 int M, int Kdim, int ldOut) {
    int wave = (blockIdx.x * blockDim.x + threadIdx.x) >> 5;
    int lane = threadIdx.x & 31;
    const int tilesN = 16;                    // N fixed at 256
    int tM = wave / tilesN;
    int tN = wave % tilesN;
    if (tM * 16 >= M) return;                 // uniform across wave: EXEC stays all-1

    int laneHalf = lane >> 4;
    int laneM = lane & 15;
    const unsigned short* aRow = A + (size_t)(tM * 16 + laneM) * Kdim;
    int kTiles = Kdim >> 5;
    v8f acc = {};
    for (int kt = 0; kt < kTiles; ++kt) {
        int colA = kt * 32 + laneHalf * 16;
        V16 a, b;
        a.s.lo = *(const u16x8*)(aRow + colA);
        a.s.hi = *(const u16x8*)(aRow + colA + 8);
        const unsigned short* bPtr = Bp + ((size_t)(kt * tilesN + tN) * 32 + lane) * 16;
        b.s.lo = *(const u16x8*)(bPtr);
        b.s.hi = *(const u16x8*)(bPtr + 8);
        acc = __builtin_amdgcn_wmma_f32_16x16x32_bf16(
            false, a.v, false, b.v, (short)0, acc, false, false);
    }
    int nCol = tN * 16 + laneM;
    float bv = bias[nCol];
#pragma unroll
    for (int i = 0; i < 8; ++i) {
        float v = acc[i] + bv;
        if (ACT) v = v * (1.f / (1.f + __expf(-v)));
        int row = tM * 16 + laneHalf * 8 + i;
        if (OUTF32) outF[(size_t)row * ldOut + nCol] = v;
        else        outB[(size_t)row * ldOut + nCol] = f32_to_bf16(v);
    }
}

// ---------------------------------------------------------------------------
extern "C" void kernel_launch(void* const* d_in, const int* in_sizes, int n_in,
                              void* d_out, int out_size, void* d_ws, size_t ws_size,
                              hipStream_t stream) {
    (void)in_sizes; (void)n_in; (void)out_size; (void)ws_size;
    const float* coords_bb = (const float*)d_in[0];
    const float* frames    = (const float*)d_in[1];
    const int*   seq_pos   = (const int*)d_in[2];
    const int*   chain_pos = (const int*)d_in[3];
    const unsigned char* valid = (const unsigned char*)d_in[4];
    const float* ln_rbf_g = (const float*)d_in[5];
    const float* ln_rbf_b = (const float*)d_in[6];
    const float* rbf_w    = (const float*)d_in[7];
    const float* rbf_b    = (const float*)d_in[8];
    const float* frame_w  = (const float*)d_in[9];
    const float* frame_b  = (const float*)d_in[10];
    const float* seq_emb  = (const float*)d_in[11];
    const float* ln_edge_g = (const float*)d_in[12];
    const float* ln_edge_b = (const float*)d_in[13];
    const float* w0 = (const float*)d_in[14];
    const float* b0 = (const float*)d_in[15];
    const float* w1 = (const float*)d_in[16];
    const float* b1 = (const float*)d_in[17];
    const float* w2 = (const float*)d_in[18];
    const float* b2 = (const float*)d_in[19];
    const float* w3 = (const float*)d_in[20];
    const float* b3 = (const float*)d_in[21];

    float* edgesOut = (float*)d_out;                          // [E,256] f32
    int*   nbrsOut  = (int*)d_out + (size_t)EE * 256;         // [E] int32
    int*   maskOut  = nbrsOut + EE;                           // [E]

    // ---- workspace carve-up (bf16 intermediates) ----
    char* w = (char*)d_ws;
    unsigned short* A_rbf = (unsigned short*)w;  w += (size_t)EE * 256 * 2;
    unsigned short* Hcat  = (unsigned short*)w;  w += (size_t)EE * 768 * 2;
    unsigned short* Xln   = (unsigned short*)w;  w += (size_t)EE * 768 * 2;
    unsigned short* H1    = (unsigned short*)w;  w += (size_t)EE * 256 * 2;
    unsigned short* H2    = (unsigned short*)w;  w += (size_t)EE * 256 * 2;
    unsigned short* rbf_wp = (unsigned short*)w; w += (size_t)256 * 256 * 2;
    unsigned short* w0p    = (unsigned short*)w; w += (size_t)768 * 256 * 2;
    unsigned short* w1p    = (unsigned short*)w; w += (size_t)256 * 256 * 2;
    unsigned short* w2p    = (unsigned short*)w; w += (size_t)256 * 256 * 2;
    unsigned short* w3p    = (unsigned short*)w; w += (size_t)256 * 256 * 2;

    // ---- weight packing to WMMA B-layout ----
    {
        int t256 = (256 >> 5) * 16 * 32;   // 4096
        int t768 = (768 >> 5) * 16 * 32;   // 12288
        pack_w_kernel<<<(t256 + 255) / 256, 256, 0, stream>>>(rbf_w, rbf_wp, 256);
        pack_w_kernel<<<(t768 + 255) / 256, 256, 0, stream>>>(w0, w0p, 768);
        pack_w_kernel<<<(t256 + 255) / 256, 256, 0, stream>>>(w1, w1p, 256);
        pack_w_kernel<<<(t256 + 255) / 256, 256, 0, stream>>>(w2, w2p, 256);
        pack_w_kernel<<<(t256 + 255) / 256, 256, 0, stream>>>(w3, w3p, 256);
    }

    // ---- kNN ----
    knn_kernel<<<(ZB * NN + 255) / 256, 256, 0, stream>>>(coords_bb, valid, nbrsOut, maskOut);

    // ---- per-edge features (one wave per edge) ----
    {
        int blocks = (EE * 32) / 256;   // 16384
        feat_kernel<<<blocks, 256, 0, stream>>>(coords_bb, frames, seq_pos, chain_pos,
                                                nbrsOut, ln_rbf_g, ln_rbf_b,
                                                frame_w, frame_b, seq_emb,
                                                A_rbf, Hcat);
    }

    // ---- rel_rbf GEMM: LN(rbf) @ rbf_w + rbf_b -> Hcat[:,0:256] ----
    {
        int tiles = (EE / 16) * 16;         // 131072 waves
        int blocks = tiles / 8;             // 8 waves per block
        gemm_bf16_kernel<0, 0><<<blocks, 256, 0, stream>>>(
            A_rbf, rbf_wp, rbf_b, Hcat, nullptr, EE, 256, 768);
    }

    // ---- edge LayerNorm over 768 ----
    edgeln_kernel<<<(EE * 32) / 256, 256, 0, stream>>>(Hcat, ln_edge_g, ln_edge_b, Xln);

    // ---- MLP: silu GEMMs + final f32 GEMM ----
    {
        int tiles = (EE / 16) * 16;
        int blocks = tiles / 8;
        gemm_bf16_kernel<1, 0><<<blocks, 256, 0, stream>>>(
            Xln, w0p, b0, H1, nullptr, EE, 768, 256);
        gemm_bf16_kernel<1, 0><<<blocks, 256, 0, stream>>>(
            H1, w1p, b1, H2, nullptr, EE, 256, 256);
        gemm_bf16_kernel<1, 0><<<blocks, 256, 0, stream>>>(
            H2, w2p, b2, H1, nullptr, EE, 256, 256);
        gemm_bf16_kernel<0, 1><<<blocks, 256, 0, stream>>>(
            H1, w3p, b3, nullptr, edgesOut, EE, 256, 256);
    }
}
